// MLA_2233382994361
// MI455X (gfx1250) — compile-verified, matrix operand order checked by reading, not run on hardware
//
#include <hip/hip_runtime.h>
#include <hip/hip_bf16.h>
#include <cmath>

typedef __hip_bfloat16 bf16;
typedef __attribute__((ext_vector_type(16))) __bf16 v16bf;
typedef __attribute__((ext_vector_type(8)))  float  v8f;

#define S_LEN    2048
#define BATCH    2
#define N_TOK    4096
#define DIM      2048
#define NHEADS   16
#define NOPE     128
#define ROPE_D   64
#define VDIM     128
#define KVRANK   512
#define DQK      576            // KVRANK + ROPE_D
#define QSTRIDE  9216           // NHEADS*DQK
#define OLSTRIDE 8192           // NHEADS*KVRANK
#define QDIM     3072           // NHEADS*(NOPE+ROPE_D)

// ---------------------------------------------------------------------------
// WMMA helpers (CDNA5 16x16x32 bf16, f32 accumulate)
// ---------------------------------------------------------------------------
union FragU { v16bf f; uint4 q[2]; };

// Load an A- or B-operand fragment from (LDS) memory laid out row-major:
// lane L<16 -> row row0+L, K-chunks [kb+8h, kb+8h+8) and [kb+16+8h, ...)
// with h = L>=16.  Two 16-byte loads per lane.  ld, kb in bf16 elements;
// row stride (ld*2 bytes) and kb must be multiples of 8 elements.
static __device__ inline v16bf load_frag(const bf16* base, int row0, int ld, int kb) {
    const int lane = threadIdx.x & 31;
    const int l    = lane & 15;
    const int hf   = lane >> 4;
    const bf16* p  = base + (size_t)(row0 + l) * ld + kb + hf * 8;
    FragU u;
    u.q[0] = *reinterpret_cast<const uint4*>(p);
    u.q[1] = *reinterpret_cast<const uint4*>(p + 16);
    return u.f;
}

static __device__ inline v8f wmma_bf16(v16bf a, v16bf b, v8f c) {
    return __builtin_amdgcn_wmma_f32_16x16x32_bf16(
        false, a, false, b, (short)0, c, false, false);
}

template <typename T> __device__ inline T to_out(float v);
template <> __device__ inline float to_out<float>(float v) { return v; }
template <> __device__ inline bf16  to_out<bf16>(float v)  { return __float2bfloat16(v); }

// ---------------------------------------------------------------------------
// Elementwise f32 -> bf16 conversion
// ---------------------------------------------------------------------------
__global__ void cvt_f32_bf16(const float* __restrict__ src, bf16* __restrict__ dst, int n) {
    int i = blockIdx.x * blockDim.x + threadIdx.x;
    int stride = gridDim.x * blockDim.x;
    for (; i < n; i += stride) dst[i] = __float2bfloat16(src[i]);
}

// Per-head transpose of the nope block of wkv_b:
// wkvbT[h][c][d] = wkv_b[h*256 + d][c], bf16.  16*512*128 elements.
__global__ void wkvb_transpose(const float* __restrict__ wkv_b, bf16* __restrict__ wkvbT) {
    int i = blockIdx.x * blockDim.x + threadIdx.x;
    const int n = NHEADS * KVRANK * NOPE;
    if (i >= n) return;
    int d = i & (NOPE - 1);
    int c = (i >> 7) & (KVRANK - 1);
    int h = i >> 16;
    wkvbT[i] = __float2bfloat16(wkv_b[((size_t)(h * 256 + d)) * KVRANK + c]);
}

// ---------------------------------------------------------------------------
// Generic tiled NT GEMM:  C(MxN) = A(MxK) * B(NxK)^T, bf16 in, f32 acc.
// 128x128 tile / block, 8 waves in a 2(m) x 4(n) grid, KC=64 per stage.
// Register-staged, double-buffered LDS pipeline: global loads for stage s+1
// are issued before the WMMA work of stage s (one barrier per stage).
// Requires K % 64 == 0 (true for all call sites: 2048, 512, 128).
// blockIdx.z selects a (head) slice via element strides aZ/bZ/cZ.
// ---------------------------------------------------------------------------
template <typename OutT>
__global__ __launch_bounds__(256) void gemm_nt(
    const bf16* __restrict__ A, long aZ, int lda,
    const bf16* __restrict__ B, long bZ, int ldb,
    OutT* __restrict__ C, long cZ, int ldc,
    int M, int N, int K)
{
    constexpr int BM = 128, BN = 128, KC = 64, LDT = KC + 8;
    __shared__ __align__(16) bf16 sA[2][BM * LDT];
    __shared__ __align__(16) bf16 sB[2][BN * LDT];

    A += (size_t)blockIdx.z * aZ;
    B += (size_t)blockIdx.z * bZ;
    C += (size_t)blockIdx.z * cZ;

    const int m0  = blockIdx.y * BM;
    const int n0  = blockIdx.x * BN;
    const int tid = threadIdx.x;
    const int wid = tid >> 5;
    const int mb  = (wid & 1) * 64;   // wave M offset in tile
    const int nb  = (wid >> 1) * 32;  // wave N offset in tile
    const int rr  = tid >> 3;         // staging row (0..31 per chunk of 256/8)
    const int cc  = (tid & 7) * 8;    // staging col offset

    uint4 ra[4], rb[4];
    auto gload = [&](int kb) {
#pragma unroll
        for (int it = 0; it < 4; ++it) {
            int r = rr + it * 32;
            ra[it] = make_uint4(0, 0, 0, 0);
            if (m0 + r < M)
                ra[it] = *reinterpret_cast<const uint4*>(A + (size_t)(m0 + r) * lda + kb + cc);
            rb[it] = make_uint4(0, 0, 0, 0);
            if (n0 + r < N)
                rb[it] = *reinterpret_cast<const uint4*>(B + (size_t)(n0 + r) * ldb + kb + cc);
        }
    };

    v8f acc[4][2];
#pragma unroll
    for (int mi = 0; mi < 4; ++mi)
#pragma unroll
        for (int ni = 0; ni < 2; ++ni) acc[mi][ni] = v8f{};

    gload(0);
    const int nstage = K / KC;
    for (int st = 0; st < nstage; ++st) {
        bf16* cA = sA[st & 1];
        bf16* cB = sB[st & 1];
#pragma unroll
        for (int it = 0; it < 4; ++it) {
            int r = rr + it * 32;
            *reinterpret_cast<uint4*>(cA + r * LDT + cc) = ra[it];
            *reinterpret_cast<uint4*>(cB + r * LDT + cc) = rb[it];
        }
        __syncthreads();
        if (st + 1 < nstage) gload((st + 1) * KC);   // overlap with compute below
#pragma unroll
        for (int kk = 0; kk < KC; kk += 32) {
            v16bf af[4], bfr[2];
#pragma unroll
            for (int mi = 0; mi < 4; ++mi) af[mi] = load_frag(cA, mb + mi * 16, LDT, kk);
#pragma unroll
            for (int ni = 0; ni < 2; ++ni) bfr[ni] = load_frag(cB, nb + ni * 16, LDT, kk);
#pragma unroll
            for (int mi = 0; mi < 4; ++mi)
#pragma unroll
                for (int ni = 0; ni < 2; ++ni)
                    acc[mi][ni] = wmma_bf16(af[mi], bfr[ni], acc[mi][ni]);
        }
    }

    // store D: element v of frag -> row = base + v + 8*(lane>=16), col = n + (lane&15)
    const int lane = tid & 31, l = lane & 15, hf = lane >> 4;
#pragma unroll
    for (int mi = 0; mi < 4; ++mi)
#pragma unroll
        for (int ni = 0; ni < 2; ++ni) {
            int cn = n0 + nb + ni * 16 + l;
            if (cn >= N) continue;
            int rb2 = m0 + mb + mi * 16 + hf * 8;
#pragma unroll
            for (int v = 0; v < 8; ++v) {
                int r = rb2 + v;
                if (r < M) C[(size_t)r * ldc + cn] = to_out<OutT>(acc[mi][ni][v]);
            }
        }
}

// ---------------------------------------------------------------------------
// Post-processing: RMSNorm(kv_c) -> keff[:, :512] and transposed cT,
// RoPE(k_pe) -> keff[:, 512:], RoPE(q_pe) -> qeff[:, h*576+512 : +576].
// One block (256 threads) per token.
// ---------------------------------------------------------------------------
__global__ __launch_bounds__(256) void mla_post(
    const bf16* __restrict__ kvb,    // (4096, 576)
    const bf16* __restrict__ qb,     // (4096, 3072)  (roped slice read here)
    const float* __restrict__ angles,// (2048, 32)
    const float* __restrict__ knw,   // (512)
    bf16* __restrict__ keff,         // (4096, 576)
    bf16* __restrict__ cT,           // (2, 512, 2048)
    bf16* __restrict__ qeff)         // (4096, 9216)
{
    const int t = blockIdx.x;
    const int b = t >> 11;
    const int s = t & (S_LEN - 1);
    const int tid = threadIdx.x;

    __shared__ float red[256];
    __shared__ float cs[64];

    const float v0 = __bfloat162float(kvb[(size_t)t * DQK + tid]);
    const float v1 = __bfloat162float(kvb[(size_t)t * DQK + 256 + tid]);
    red[tid] = v0 * v0 + v1 * v1;
    __syncthreads();
#pragma unroll
    for (int st = 128; st > 0; st >>= 1) {
        if (tid < st) red[tid] += red[tid + st];
        __syncthreads();
    }
    const float inv = rsqrtf(red[0] * (1.0f / (float)KVRANK) + 1e-6f);

    {
        float c0 = v0 * inv * knw[tid];
        float c1 = v1 * inv * knw[tid + 256];
        bf16 b0 = __float2bfloat16(c0), b1 = __float2bfloat16(c1);
        keff[(size_t)t * DQK + tid]       = b0;
        keff[(size_t)t * DQK + 256 + tid] = b1;
        cT[((size_t)b * KVRANK + tid) * S_LEN + s]       = b0;
        cT[((size_t)b * KVRANK + tid + 256) * S_LEN + s] = b1;
    }

    if (tid < 32) {
        float a = angles[(size_t)s * 32 + tid];
        cs[tid]      = __cosf(a);
        cs[32 + tid] = __sinf(a);
    }
    __syncthreads();

    // k_pe rope -> keff[:, 512:576]
    if (tid < 32) {
        float xr = __bfloat162float(kvb[(size_t)t * DQK + KVRANK + 2 * tid]);
        float xi = __bfloat162float(kvb[(size_t)t * DQK + KVRANK + 2 * tid + 1]);
        float c = cs[tid], sn = cs[32 + tid];
        keff[(size_t)t * DQK + KVRANK + 2 * tid]     = __float2bfloat16(xr * c - xi * sn);
        keff[(size_t)t * DQK + KVRANK + 2 * tid + 1] = __float2bfloat16(xr * sn + xi * c);
    }

    // q_pe rope: 16 heads x 32 pairs = 512 pairs
#pragma unroll
    for (int j = 0; j < 2; ++j) {
        int pid = tid + j * 256;
        int h = pid >> 5, pp = pid & 31;
        float xr = __bfloat162float(qb[(size_t)t * QDIM + h * 192 + NOPE + 2 * pp]);
        float xi = __bfloat162float(qb[(size_t)t * QDIM + h * 192 + NOPE + 2 * pp + 1]);
        float c = cs[pp], sn = cs[32 + pp];
        qeff[(size_t)t * QSTRIDE + h * DQK + KVRANK + 2 * pp]     = __float2bfloat16(xr * c - xi * sn);
        qeff[(size_t)t * QSTRIDE + h * DQK + KVRANK + 2 * pp + 1] = __float2bfloat16(xr * sn + xi * c);
    }
}

// ---------------------------------------------------------------------------
// Flash attention over the latent space.
// Block = (b, h, 64-query tile), 8 waves.  Q tile LDS-resident; per key tile
// stage K (64x576) and V^T (512x64) in LDS.  S (f32) and P (bf16) in LDS.
// Wave roles: QK:   (wid>>1) -> 16 query rows, (wid&1) -> 32 key cols
//             PV:   (wid>>1) -> 16 query rows, (wid&1) -> 256 latent cols
// QK and PV inner loops are software-pipelined (fragments loaded ahead of
// their consuming WMMA) so the matrix pipe is not serialized on DS latency.
// ---------------------------------------------------------------------------
__global__ __launch_bounds__(256) void mla_attn(
    const bf16* __restrict__ qeff,   // (4096, 9216)
    const bf16* __restrict__ keff,   // (4096, 576)
    const bf16* __restrict__ cT,     // (2, 512, 2048)
    bf16* __restrict__ olat,         // (4096, 8192)
    float scale)
{
    constexpr int BM = 64, BN = 64;
    constexpr int QLD = DQK + 8;     // 584
    constexpr int KLD = DQK + 8;
    constexpr int VLD = BN + 8;      // 72
    constexpr int SLD = BN + 4;      // 68
    constexpr int PLD = BN + 8;      // 72

    extern __shared__ __align__(16) char smem[];
    bf16*  sQ   = reinterpret_cast<bf16*>(smem);
    bf16*  sK   = sQ + BM * QLD;
    bf16*  sV   = sK + BN * KLD;                 // V^T : 512 rows x 64 keys
    float* sS   = reinterpret_cast<float*>(sV + KVRANK * VLD);
    bf16*  sP   = reinterpret_cast<bf16*>(sS + BM * SLD);
    float* rowM = reinterpret_cast<float*>(sP + BM * PLD);
    float* rowL = rowM + BM;
    float* rowC = rowL + BM;

    const int b = blockIdx.z, h = blockIdx.y, it = blockIdx.x;
    const int tid = threadIdx.x, lane = tid & 31, wid = tid >> 5;
    const int l = lane & 15, hf = lane >> 4;
    const int mb = (wid >> 1) * 16;
    const int wx = wid & 1;
    const size_t qrow0 = (size_t)(b * S_LEN + it * BM);

    // Load Q tile (64 x 576)
    for (int i = tid; i < BM * DQK / 8; i += 256) {
        int r = i / (DQK / 8), cc = (i % (DQK / 8)) * 8;
        *reinterpret_cast<uint4*>(sQ + r * QLD + cc) =
            *reinterpret_cast<const uint4*>(qeff + (qrow0 + r) * QSTRIDE + h * DQK + cc);
    }
    if (tid < BM) { rowM[tid] = -3.0e38f; rowL[tid] = 0.0f; }

    v8f oacc[16];
#pragma unroll
    for (int ni = 0; ni < 16; ++ni) oacc[ni] = v8f{};

    for (int jt = 0; jt <= it; ++jt) {
        __syncthreads();   // previous PV done before restaging K/V^T
        const size_t krow0 = (size_t)(b * S_LEN + jt * BN);
        for (int i = tid; i < BN * DQK / 8; i += 256) {
            int r = i / (DQK / 8), cc = (i % (DQK / 8)) * 8;
            *reinterpret_cast<uint4*>(sK + r * KLD + cc) =
                *reinterpret_cast<const uint4*>(keff + (krow0 + r) * DQK + cc);
        }
        for (int i = tid; i < KVRANK * BN / 8; i += 256) {
            int r = i / (BN / 8), cc = (i % (BN / 8)) * 8;
            *reinterpret_cast<uint4*>(sV + r * VLD + cc) =
                *reinterpret_cast<const uint4*>(cT + ((size_t)b * KVRANK + r) * S_LEN + jt * BN + cc);
        }
        // Prefetch next key tile toward the WGP while this one computes
        // (cacheline granular; keff rows are 1152B = 9 lines, cT rows 1 line).
        if (jt + 1 <= it) {
            const size_t krow1 = (size_t)(b * S_LEN + (jt + 1) * BN);
            for (int i = tid; i < BN * 9; i += 256)
                __builtin_prefetch(keff + (krow1 + i / 9) * DQK + (i % 9) * 64, 0, 3);
            for (int i = tid; i < KVRANK; i += 256)
                __builtin_prefetch(cT + ((size_t)b * KVRANK + i) * S_LEN + (jt + 1) * BN, 0, 3);
        }
        __syncthreads();

        // ---- S = Q K^T (16 rows x 32 cols per wave), pipelined ----
        const int nb = wx * 32;
        v8f sf[2] = { v8f{}, v8f{} };
        {
            v16bf aq  = load_frag(sQ, mb, QLD, 0);
            v16bf bk0 = load_frag(sK, nb, KLD, 0);
            v16bf bk1 = load_frag(sK, nb + 16, KLD, 0);
            for (int k = 0; k < DQK; k += 32) {
                const int kn = (k + 32 < DQK) ? k + 32 : 0;   // wrap: harmless extra load
                v16bf aqn  = load_frag(sQ, mb, QLD, kn);
                v16bf bk0n = load_frag(sK, nb, KLD, kn);
                v16bf bk1n = load_frag(sK, nb + 16, KLD, kn);
                sf[0] = wmma_bf16(aq, bk0, sf[0]);
                sf[1] = wmma_bf16(aq, bk1, sf[1]);
                aq = aqn; bk0 = bk0n; bk1 = bk1n;
            }
        }
        // scale + causal mask, write S to LDS
#pragma unroll
        for (int ni = 0; ni < 2; ++ni)
#pragma unroll
            for (int v = 0; v < 8; ++v) {
                int r = mb + hf * 8 + v;
                int cidx = nb + ni * 16 + l;
                int qpos = it * BM + r, kpos = jt * BN + cidx;
                sS[r * SLD + cidx] = (kpos <= qpos) ? sf[ni][v] * scale : -3.0e38f;
            }
        __syncthreads();

        // ---- online softmax stats: wave owns rows wid*8 .. wid*8+7 ----
#pragma unroll
        for (int rr = 0; rr < 8; ++rr) {
            int r = wid * 8 + rr;
            float x0 = sS[r * SLD + lane];
            float x1 = sS[r * SLD + lane + 32];
            float mx = fmaxf(x0, x1);
#pragma unroll
            for (int o = 16; o > 0; o >>= 1) mx = fmaxf(mx, __shfl_xor(mx, o, 32));
            float mold = rowM[r];
            float mnew = fmaxf(mold, mx);
            float p0 = __expf(x0 - mnew);
            float p1 = __expf(x1 - mnew);
            float sm = p0 + p1;
#pragma unroll
            for (int o = 16; o > 0; o >>= 1) sm += __shfl_xor(sm, o, 32);
            sP[r * PLD + lane]      = __float2bfloat16(p0);
            sP[r * PLD + lane + 32] = __float2bfloat16(p1);
            if (lane == 0) {
                float corr = __expf(mold - mnew);
                rowM[r] = mnew;
                rowC[r] = corr;
                rowL[r] = rowL[r] * corr + sm;
            }
        }
        __syncthreads();

        // ---- rescale running O, then O += P V (2-deep pipelined B-frags) ----
        float cr[8];
        {
            const float* cp = rowC + mb + hf * 8;
#pragma unroll
            for (int v = 0; v < 8; ++v) cr[v] = cp[v];
        }
#pragma unroll
        for (int ni = 0; ni < 16; ++ni)
#pragma unroll
            for (int v = 0; v < 8; ++v) oacc[ni][v] *= cr[v];

        const int vb = wx * 256;
        {
            v16bf pa0 = load_frag(sP, mb, PLD, 0);
            v16bf pa1 = load_frag(sP, mb, PLD, 32);
            v16bf bv0 = load_frag(sV, vb, VLD, 0);
            v16bf bv1 = load_frag(sV, vb + 16, VLD, 0);
#pragma unroll
            for (int s = 0; s < 32; ++s) {
                const int ni = s & 15;
                const int sn = s + 2;
                v16bf bnew = load_frag(sV, vb + (sn & 15) * 16, VLD, ((sn >> 4) & 1) * 32);
                if ((s & 1) == 0) {
                    oacc[ni] = wmma_bf16((s < 16) ? pa0 : pa1, bv0, oacc[ni]);
                    bv0 = bnew;
                } else {
                    oacc[ni] = wmma_bf16((s < 16) ? pa0 : pa1, bv1, oacc[ni]);
                    bv1 = bnew;
                }
            }
        }
    }

    // ---- epilogue: normalize by 1/l and store bf16 ----
    float li[8];
    {
        const float* lp = rowL + mb + hf * 8;
#pragma unroll
        for (int v = 0; v < 8; ++v) li[v] = 1.0f / lp[v];
    }
#pragma unroll
    for (int ni = 0; ni < 16; ++ni)
#pragma unroll
        for (int v = 0; v < 8; ++v) {
            int r = mb + hf * 8 + v;
            int c = wx * 256 + ni * 16 + l;
            olat[(qrow0 + r) * OLSTRIDE + h * KVRANK + c] =
                __float2bfloat16(oacc[ni][v] * li[v]);
        }
}

// ---------------------------------------------------------------------------
// Host launcher
// ---------------------------------------------------------------------------
extern "C" void kernel_launch(void* const* d_in, const int* in_sizes, int n_in,
                              void* d_out, int out_size, void* d_ws, size_t ws_size,
                              hipStream_t stream) {
    (void)in_sizes; (void)n_in; (void)out_size; (void)ws_size;
    const float* x      = (const float*)d_in[0];
    const float* angles = (const float*)d_in[1];
    const float* wq     = (const float*)d_in[2];
    const float* wkv_a  = (const float*)d_in[3];
    const float* wkv_b  = (const float*)d_in[4];
    const float* wo     = (const float*)d_in[5];
    const float* knw    = (const float*)d_in[6];
    float* out = (float*)d_out;

    // ---- carve workspace ----
    char* p = (char*)d_ws;
    auto carve = [&](size_t elems) {
        char* r = p;
        p += ((elems * sizeof(bf16)) + 255) & ~(size_t)255;
        return (bf16*)r;
    };
    bf16* xb    = carve((size_t)N_TOK * DIM);
    bf16* wqb   = carve((size_t)QDIM * DIM);
    bf16* wkvab = carve((size_t)DQK * DIM);
    bf16* wkvbb = carve((size_t)NHEADS * 256 * KVRANK);
    bf16* wkvbT = carve((size_t)NHEADS * KVRANK * NOPE);
    bf16* wob   = carve((size_t)DIM * DIM);
    bf16* qb    = carve((size_t)N_TOK * QDIM);
    bf16* kvb   = carve((size_t)N_TOK * DQK);
    bf16* qeff  = carve((size_t)N_TOK * QSTRIDE);
    bf16* keff  = carve((size_t)N_TOK * DQK);
    bf16* cT    = carve((size_t)BATCH * KVRANK * S_LEN);
    bf16* olat  = carve((size_t)N_TOK * OLSTRIDE);
    bf16* obuf  = carve((size_t)N_TOK * DIM);

    const double msc = 0.1 * std::log(40.0) + 1.0;
    const float scale = (float)((msc * msc) / std::sqrt((double)(NOPE + ROPE_D)));

    // ---- 1. precision conversion ----
    cvt_f32_bf16<<<4096, 256, 0, stream>>>(x,     xb,    N_TOK * DIM);
    cvt_f32_bf16<<<4096, 256, 0, stream>>>(wq,    wqb,   QDIM * DIM);
    cvt_f32_bf16<<<2048, 256, 0, stream>>>(wkv_a, wkvab, DQK * DIM);
    cvt_f32_bf16<<<2048, 256, 0, stream>>>(wkv_b, wkvbb, NHEADS * 256 * KVRANK);
    cvt_f32_bf16<<<4096, 256, 0, stream>>>(wo,    wob,   DIM * DIM);
    wkvb_transpose<<<(NHEADS * KVRANK * NOPE + 255) / 256, 256, 0, stream>>>(wkv_b, wkvbT);

    // ---- 2. q = x @ wq^T  (4096x3072, K=2048) ----
    gemm_nt<bf16><<<dim3(QDIM / 128, N_TOK / 128, 1), 256, 0, stream>>>(
        xb, 0, DIM, wqb, 0, DIM, qb, 0, QDIM, N_TOK, QDIM, DIM);

    // ---- 3. kv = x @ wkv_a^T  (4096x576) ----
    gemm_nt<bf16><<<dim3((DQK + 127) / 128, N_TOK / 128, 1), 256, 0, stream>>>(
        xb, 0, DIM, wkvab, 0, DIM, kvb, 0, DQK, N_TOK, DQK, DIM);

    // ---- 4. rmsnorm + rope -> keff, cT, qeff(pe) ----
    mla_post<<<N_TOK, 256, 0, stream>>>(kvb, qb, angles, knw, keff, cT, qeff);

    // ---- 5. q_abs = q_nope @ wkvbT^T per head -> qeff[:, h*576 : h*576+512] ----
    gemm_nt<bf16><<<dim3(KVRANK / 128, N_TOK / 128, NHEADS), 256, 0, stream>>>(
        qb, 192, QDIM, wkvbT, (long)KVRANK * NOPE, NOPE,
        qeff, DQK, QSTRIDE, N_TOK, KVRANK, NOPE);

    // ---- 6. flash attention in latent space -> olat ----
    {
        constexpr size_t SMEM =
            (size_t)(64 * (DQK + 8)) * 2 * sizeof(bf16) +   // sQ + sK
            (size_t)(KVRANK * (64 + 8)) * sizeof(bf16) +    // sV (V^T)
            (size_t)(64 * (64 + 4)) * sizeof(float) +       // sS
            (size_t)(64 * (64 + 8)) * sizeof(bf16) +        // sP
            (size_t)(3 * 64) * sizeof(float);               // rowM/L/C
        mla_attn<<<dim3(S_LEN / 64, NHEADS, BATCH), 256, SMEM, stream>>>(
            qeff, keff, cT, olat, scale);
    }

    // ---- 7. o = olat @ wkv_b_v^T per head -> obuf[:, h*128 : +128] ----
    gemm_nt<bf16><<<dim3(1, N_TOK / 128, NHEADS), 256, 0, stream>>>(
        olat, KVRANK, OLSTRIDE,
        wkvbb + (size_t)NOPE * KVRANK, (long)256 * KVRANK, KVRANK,
        obuf, VDIM, DIM, N_TOK, VDIM, KVRANK);

    // ---- 8. out = obuf @ wo^T  (4096x2048, f32) ----
    gemm_nt<float><<<dim3(DIM / 128, N_TOK / 128, 1), 256, 0, stream>>>(
        obuf, 0, DIM, wob, 0, DIM, out, 0, DIM, N_TOK, DIM, DIM);
}
